// PointerGenerator_67388036874628
// MI455X (gfx1250) — matrix operand order repaired and checked by reading, not compiled, not run on hardware
//
#include <hip/hip_runtime.h>
#include <math.h>

// ---------------------------------------------------------------------------
// Pointer-generator decoder step for MI455X (gfx1250, wave32, WMMA).
//
// B=4 T=256 S=400 D=512 V=50000 NX=50 VEXT=50050, M = B*T = 1024 rows.
// Heavy op: logits = x @ fc_w^T  (1024 x 50000, K=512) ~52 GFLOP.
//   - bf16 WMMA (v_wmma_f32_16x16x32_bf16), f32 accumulate.
//   - fc_w bf16 (51 MB) fits in the 192 MB L2 -> after first sweep the GEMM is
//     WMMA-bound, so the K-loop is double-buffered (BK=64, 1 barrier/step)
//     with async global->LDS copies (ASYNCcnt) to overlap streaming with math.
//   - softmax fused as pass1 (sum-exp reduction) + pass2 (recompute & write
//     p_gen * exp/sum): cheaper than 400 MB of logit spill traffic.
// ---------------------------------------------------------------------------

#define CB 4
#define CT 256
#define CS 400
#define CD 512
#define CV 50000
#define CNX 50
#define CVEXT 50050
#define CM 1024      // B*T
#define CKR 1600     // B*S rows of k

// gfx1250 async global->LDS copies via inline asm (portable across ROCm 7.2 /
// amdgpu-toolchain per bridge doc). Flip to 0 to fall back to reg staging.
#define USE_ASYNC_LDS 1

typedef __attribute__((ext_vector_type(4)))  unsigned int uint4v;
typedef __attribute__((ext_vector_type(8)))  float        v8f;
typedef __attribute__((ext_vector_type(16))) __bf16       v16bf;

union FragBF { uint4v q[2]; v16bf v; };

__device__ __forceinline__ unsigned short f2bf(float f) {
  unsigned int u = __float_as_uint(f);
  u += 0x7FFFu + ((u >> 16) & 1u);               // round-to-nearest-even
  return (unsigned short)(u >> 16);
}

__device__ __forceinline__ unsigned lds_addr32(const void* p) {
  // generic pointer to LDS: low 32 bits are the LDS byte address
  return (unsigned)(unsigned long long)p;
}

__device__ __forceinline__ void async_copy_b128(unsigned lds, const void* g) {
#if USE_ASYNC_LDS
  asm volatile("global_load_async_to_lds_b128 %0, %1, off"
               :: "v"(lds), "v"(g) : "memory");
#else
  (void)lds; (void)g;
#endif
}

__device__ __forceinline__ void wait_async0() {
#if USE_ASYNC_LDS
  asm volatile("s_wait_asynccnt 0" ::: "memory");
#endif
}

// --------------------------- elementwise helpers ---------------------------

__global__ void f32_to_bf16_kernel(const float* __restrict__ src,
                                   unsigned short* __restrict__ dst, int n) {
  int i = blockIdx.x * blockDim.x + threadIdx.x;
  int stride = gridDim.x * blockDim.x;
  for (; i < n; i += stride) dst[i] = f2bf(src[i]);
}

__global__ void fill_kernel(float* __restrict__ p, int n, float v) {
  int i = blockIdx.x * blockDim.x + threadIdx.x;
  int stride = gridDim.x * blockDim.x;
  for (; i < n; i += stride) p[i] = v;
}

// zero out[:, :, V:VEXT] (extended-vocab tail), 1024*50 elements
__global__ void tailzero_kernel(float* __restrict__ out) {
  int i = blockIdx.x * 256 + threadIdx.x;
  if (i < CM * CNX) {
    int row = i / CNX, c = i % CNX;
    out[(size_t)row * CVEXT + CV + c] = 0.0f;
  }
}

__global__ void log_kernel(float* __restrict__ out, long long n) {
  long long i = (long long)blockIdx.x * blockDim.x + threadIdx.x;
  long long stride = (long long)gridDim.x * blockDim.x;
  for (; i < n; i += stride) out[i] = __logf(fmaxf(out[i], 1e-9f));
}

// ----------------------------- p_gen (sigmoid) -----------------------------

__global__ __launch_bounds__(256) void pgen_kernel(
    const float* __restrict__ x, const float* __restrict__ pw,
    const float* __restrict__ pb, float* __restrict__ pgen) {
  int w = threadIdx.x >> 5, lane = threadIdx.x & 31;
  int row = blockIdx.x * 8 + w;                   // 128 blocks * 8 waves
  if (row >= CM) return;
  const float* xr = x + (size_t)row * CD;
  float s = 0.0f;
  for (int i = lane; i < CD; i += 32) s += xr[i] * pw[i];
  for (int off = 16; off > 0; off >>= 1) s += __shfl_xor(s, off, 32);
  if (lane == 0) pgen[row] = 1.0f / (1.0f + __expf(-(s + pb[0])));
}

// -------------------- q/k projection GEMM (WMMA bf16) ----------------------
// Rows 0..1023: q = x @ wq^T + bq ; rows 1024..2623: k = enc @ wk^T + bk.
// BM=64, BN=128, BK=32, 8 waves (2 m x 4 n), wave tile 32x32.

__global__ __launch_bounds__(256) void qk_gemm(
    const unsigned short* __restrict__ xbf,   // 1024x512
    const unsigned short* __restrict__ encbf, // 1600x512
    const unsigned short* __restrict__ wqbf,  // 512x512
    const unsigned short* __restrict__ wkbf,  // 512x512
    const float* __restrict__ bq, const float* __restrict__ bk,
    unsigned short* __restrict__ qbf,         // 1024x512
    unsigned short* __restrict__ kbf) {       // 1600x512
  const int R0 = blockIdx.y * 64;
  const bool isQ = (R0 < CM);
  const unsigned short* A = isQ ? xbf : encbf;
  const unsigned short* W = isQ ? wqbf : wkbf;
  const float* bias = isQ ? bq : bk;
  unsigned short* dst = isQ ? qbf : kbf;
  const int r0 = isQ ? R0 : (R0 - CM);
  const int n0 = blockIdx.x * 128;

  const int tid = threadIdx.x;
  const int lane = tid & 31, w = tid >> 5;
  const int wm = w & 1, wn = w >> 1;
  const int lm = lane & 15, half = lane >> 4;

  __shared__ unsigned short sA[64 * 40];
  __shared__ unsigned short sB[128 * 40];

  v8f cc[2][2] = {};

  for (int k0 = 0; k0 < CD; k0 += 32) {
    __syncthreads();
    {                                           // A tile: 64 rows x 32 halves
      int r = tid >> 2, c = tid & 3;
      *reinterpret_cast<uint4v*>(&sA[r * 40 + c * 8]) =
          *reinterpret_cast<const uint4v*>(A + (size_t)(r0 + r) * CD + k0 + c * 8);
    }
#pragma unroll
    for (int i = 0; i < 2; i++) {               // B tile: 128 rows x 32 halves
      int chunk = tid + i * 256;
      int r = chunk >> 2, c = chunk & 3;
      *reinterpret_cast<uint4v*>(&sB[r * 40 + c * 8]) =
          *reinterpret_cast<const uint4v*>(W + (size_t)(n0 + r) * CD + k0 + c * 8);
    }
    __syncthreads();

    FragBF a[2], bb[2];
    const int akb = half * 8;
#pragma unroll
    for (int mi = 0; mi < 2; mi++) {
      int row = wm * 32 + mi * 16 + lm;
      a[mi].q[0] = *reinterpret_cast<const uint4v*>(&sA[row * 40 + akb]);
      a[mi].q[1] = *reinterpret_cast<const uint4v*>(&sA[row * 40 + akb + 16]);
    }
    const int bkb = half * 8;
#pragma unroll
    for (int ni = 0; ni < 2; ni++) {
      int col = wn * 32 + ni * 16 + lm;
      bb[ni].q[0] = *reinterpret_cast<const uint4v*>(&sB[col * 40 + bkb]);
      bb[ni].q[1] = *reinterpret_cast<const uint4v*>(&sB[col * 40 + bkb + 16]);
    }
#pragma unroll
    for (int mi = 0; mi < 2; mi++)
#pragma unroll
      for (int ni = 0; ni < 2; ni++)
        cc[mi][ni] = __builtin_amdgcn_wmma_f32_16x16x32_bf16(
            false, a[mi].v, false, bb[ni].v, (short)0, cc[mi][ni], false, false);
  }

#pragma unroll
  for (int mi = 0; mi < 2; mi++) {
    int rowbase = r0 + wm * 32 + mi * 16 + half * 8;
#pragma unroll
    for (int ni = 0; ni < 2; ni++) {
      int n = n0 + wn * 32 + ni * 16 + lm;
      float bv = bias[n];
#pragma unroll
      for (int r = 0; r < 8; r++)
        dst[(size_t)(rowbase + r) * CD + n] = f2bf(cc[mi][ni][r] + bv);
    }
  }
}

// ----------------- attention scores + softmax + attn_dist -------------------
// One block per (b, 16-row t tile). scores = q.k / sqrt(D), mask, softmax(S),
// attn = (1 - p_gen) * probs.

__global__ __launch_bounds__(256) void attn_kernel(
    const unsigned short* __restrict__ qbf,  // 1024x512
    const unsigned short* __restrict__ kbf,  // 1600x512
    const int* __restrict__ src_mask,        // B x S
    const float* __restrict__ pgen,          // 1024
    float* __restrict__ attn) {              // 1024 x 400
  const int blk = blockIdx.x;                // 64 blocks
  const int b = blk >> 4;
  const int t0 = (blk & 15) * 16;
  const int tid = threadIdx.x, lane = tid & 31, w = tid >> 5;
  const int lm = lane & 15, half = lane >> 4;

  __shared__ unsigned short sQ[16 * 520];    // 16 rows x 512 (+pad)
  __shared__ float sS[16 * 416];             // scores, S padded to 26 tiles

#pragma unroll
  for (int i = 0; i < 4; i++) {              // stage q tile (16 KB)
    int chunk = tid + i * 256;               // 1024 chunks of 16B
    int r = chunk >> 6, c = chunk & 63;
    *reinterpret_cast<uint4v*>(&sQ[r * 520 + c * 8]) =
        *reinterpret_cast<const uint4v*>(qbf + (size_t)(b * CT + t0 + r) * CD + c * 8);
  }
  __syncthreads();

  const float scale = 0.044194173824159216f; // 1/sqrt(512)
  for (int st = w; st < 26; st += 8) {       // 26 s-tiles of 16
    v8f cc = {};
    int s = st * 16 + lm;
    int srow = (s < CS) ? (b * CS + s) : (b * CS);
    const unsigned short* kb = kbf + (size_t)srow * CD;
#pragma unroll
    for (int k0 = 0; k0 < CD; k0 += 32) {
      FragBF a, bb;
      a.q[0] = *reinterpret_cast<const uint4v*>(&sQ[lm * 520 + k0 + half * 8]);
      a.q[1] = *reinterpret_cast<const uint4v*>(&sQ[lm * 520 + k0 + half * 8 + 16]);
      bb.q[0] = *reinterpret_cast<const uint4v*>(kb + k0 + half * 8);
      bb.q[1] = *reinterpret_cast<const uint4v*>(kb + k0 + half * 8 + 16);
      cc = __builtin_amdgcn_wmma_f32_16x16x32_bf16(
          false, a.v, false, bb.v, (short)0, cc, false, false);
    }
    int col = st * 16 + lm;
    bool valid = (col < CS) && (src_mask[b * CS + col] != 0);
#pragma unroll
    for (int r = 0; r < 8; r++) {
      int m = half * 8 + r;
      sS[m * 416 + col] = valid ? cc[r] * scale : -1.0e9f;
    }
  }
  __syncthreads();

  for (int rr = 0; rr < 2; rr++) {           // 8 waves x 2 rows = 16 rows
    int row = w * 2 + rr;
    float mx = -3.0e38f;
    for (int c = lane; c < 416; c += 32) mx = fmaxf(mx, sS[row * 416 + c]);
    for (int off = 16; off > 0; off >>= 1) mx = fmaxf(mx, __shfl_xor(mx, off, 32));
    float sum = 0.0f;
    for (int c = lane; c < 416; c += 32) sum += __expf(sS[row * 416 + c] - mx);
    for (int off = 16; off > 0; off >>= 1) sum += __shfl_xor(sum, off, 32);
    int grow = b * CT + t0 + row;
    float coef = (1.0f - pgen[grow]) / sum;
    float* dst = attn + (size_t)grow * CS;
    for (int c = lane; c < CS; c += 32)
      dst[c] = __expf(sS[row * 416 + c] - mx) * coef;
  }
}

// ------------------- vocab GEMM (the 52-GFLOP workhorse) --------------------
// M=1024, N=50000, K=512. BM=128, BN=128, BK=64; 8 waves (2 m x 4 n),
// wave tile 64x32 (4x2 subtiles, 64 accum VGPRs).
// Double-buffered LDS (one barrier per K-step); next tile streamed with
// global_load_async_to_lds_b128 (ASYNCcnt) while WMMAs consume current tile.
// LDS rows padded to 72 halves (144B) -> conflict-free b128 fragment reads.
// PASS 1: accumulate sum-exp per row (LDS ds atomics + one global atomic).
// PASS 2: recompute, write out = p_gen * exp(logit) / sumexp.

template <int PASS>
__global__ __launch_bounds__(256) void vocab_gemm(
    const unsigned short* __restrict__ Xbf,  // 1024x512
    const unsigned short* __restrict__ Wbf,  // 50000x512
    const float* __restrict__ fc_b,          // V
    const float* __restrict__ pgen,          // 1024
    float* __restrict__ sumexp,              // 1024
    float* __restrict__ out) {               // 1024 x VEXT
  const int m0 = blockIdx.y * 128;
  const int n0 = blockIdx.x * 128;
  const int tid = threadIdx.x;
  const int lane = tid & 31, w = tid >> 5;
  const int wm = w & 1, wn = w >> 1;
  const int lm = lane & 15, half = lane >> 4;

  __shared__ unsigned short sA[2][128 * 72];
  __shared__ unsigned short sB[2][128 * 72];
  __shared__ float rowsum[128];
  if (PASS == 1 && tid < 128) rowsum[tid] = 0.0f;

  // Per-thread staging chunks: 1024 chunks of 16B per 128x64 tile, 4/thread.
  int crow[4], ccol[4], crn[4];
#pragma unroll
  for (int i = 0; i < 4; i++) {
    int chunk = tid + i * 256;
    crow[i] = chunk >> 3;                  // 0..127
    ccol[i] = (chunk & 7) * 8;             // halves 0..56
    int rn = n0 + crow[i]; if (rn >= CV) rn = CV - 1;
    crn[i] = rn;
  }

  uint4v ra[4], rb[4];                     // staging regs (sync fallback)

  auto issue_loads = [&](int k0, int buf) {
#pragma unroll
    for (int i = 0; i < 4; i++) {
      const unsigned short* ga = Xbf + (size_t)(m0 + crow[i]) * CD + k0 + ccol[i];
      const unsigned short* gb = Wbf + (size_t)crn[i] * CD + k0 + ccol[i];
#if USE_ASYNC_LDS
      async_copy_b128(lds_addr32(&sA[buf][crow[i] * 72 + ccol[i]]), ga);
      async_copy_b128(lds_addr32(&sB[buf][crow[i] * 72 + ccol[i]]), gb);
#else
      ra[i] = *reinterpret_cast<const uint4v*>(ga);
      rb[i] = *reinterpret_cast<const uint4v*>(gb);
#endif
    }
    (void)buf;
  };
  auto commit_loads = [&](int buf) {
#if !USE_ASYNC_LDS
#pragma unroll
    for (int i = 0; i < 4; i++) {
      *reinterpret_cast<uint4v*>(&sA[buf][crow[i] * 72 + ccol[i]]) = ra[i];
      *reinterpret_cast<uint4v*>(&sB[buf][crow[i] * 72 + ccol[i]]) = rb[i];
    }
#else
    (void)buf;
#endif
  };

  v8f cc[4][2] = {};

  // prologue: tile 0 -> buffer 0
  issue_loads(0, 0);
  commit_loads(0);
  wait_async0();
  __syncthreads();

  for (int ks = 0; ks < 8; ks++) {         // 8 K-steps of 64
    const int cur = ks & 1;
    if (ks < 7) issue_loads((ks + 1) * 64, 1 - cur);

#pragma unroll
    for (int kk = 0; kk < 64; kk += 32) {  // two WMMA sub-steps per tile
      FragBF a[4], bb[2];
      const int akb = kk + half * 8;
#pragma unroll
      for (int mi = 0; mi < 4; mi++) {
        int row = wm * 64 + mi * 16 + lm;
        a[mi].q[0] = *reinterpret_cast<const uint4v*>(&sA[cur][row * 72 + akb]);
        a[mi].q[1] = *reinterpret_cast<const uint4v*>(&sA[cur][row * 72 + akb + 16]);
      }
      const int bkb = kk + half * 8;
#pragma unroll
      for (int ni = 0; ni < 2; ni++) {
        int col = wn * 32 + ni * 16 + lm;
        bb[ni].q[0] = *reinterpret_cast<const uint4v*>(&sB[cur][col * 72 + bkb]);
        bb[ni].q[1] = *reinterpret_cast<const uint4v*>(&sB[cur][col * 72 + bkb + 16]);
      }
#pragma unroll
      for (int mi = 0; mi < 4; mi++)
#pragma unroll
        for (int ni = 0; ni < 2; ni++)
          cc[mi][ni] = __builtin_amdgcn_wmma_f32_16x16x32_bf16(
              false, a[mi].v, false, bb[ni].v, (short)0, cc[mi][ni], false, false);
    }

    if (ks < 7) commit_loads(1 - cur);
    wait_async0();                         // own async copies landed in LDS
    __syncthreads();                       // whole tile visible to block
  }

#pragma unroll
  for (int mi = 0; mi < 4; mi++) {
    const int rbase = wm * 64 + mi * 16 + half * 8;   // row within block tile
    float pg[8], inv[8];
    if (PASS == 2) {
#pragma unroll
      for (int r = 0; r < 8; r++) {
        pg[r] = pgen[m0 + rbase + r];
        inv[r] = 1.0f / sumexp[m0 + rbase + r];
      }
    }
#pragma unroll
    for (int ni = 0; ni < 2; ni++) {
      int n = n0 + wn * 32 + ni * 16 + lm;
      bool nval = n < CV;
      float bias = nval ? fc_b[n] : 0.0f;
      if (PASS == 1) {
#pragma unroll
        for (int r = 0; r < 8; r++) {
          float e = nval ? __expf(cc[mi][ni][r] + bias) : 0.0f;
          e += __shfl_xor(e, 1, 32);
          e += __shfl_xor(e, 2, 32);
          e += __shfl_xor(e, 4, 32);
          e += __shfl_xor(e, 8, 32);
          if (lm == 0) atomicAdd(&rowsum[rbase + r], e);
        }
      } else if (nval) {
#pragma unroll
        for (int r = 0; r < 8; r++) {
          float p = pg[r] * __expf(cc[mi][ni][r] + bias) * inv[r];
          out[(size_t)(m0 + rbase + r) * CVEXT + n] = p;
        }
      }
    }
  }

  if (PASS == 1) {
    __syncthreads();
    if (tid < 128) atomicAdd(&sumexp[m0 + tid], rowsum[tid]);
  }
}

// -------------------------- scatter-add (copy dist) -------------------------

__global__ void scatter_kernel(const int* __restrict__ ext_idx,   // B x S
                               const float* __restrict__ attn,    // 1024 x 400
                               float* __restrict__ out) {
  int i = blockIdx.x * 256 + threadIdx.x;                          // 409600
  if (i >= CM * CS) return;
  int s = i % CS;
  int bt = i / CS;
  int b = bt >> 8;                                                 // T = 256
  int idx = ext_idx[b * CS + s];
  if (idx < 0) idx = 0;
  if (idx >= CVEXT) idx = CVEXT - 1;
  atomicAdd(out + (size_t)bt * CVEXT + idx, attn[i]);
}

// --------------------------------- launch ----------------------------------

extern "C" void kernel_launch(void* const* d_in, const int* in_sizes, int n_in,
                              void* d_out, int out_size, void* d_ws, size_t ws_size,
                              hipStream_t stream) {
  (void)in_sizes; (void)n_in; (void)out_size; (void)ws_size;

  const float* x        = (const float*)d_in[0];   // B,T,D
  const float* enc      = (const float*)d_in[1];   // B,S,D
  const int*   src_mask = (const int*)d_in[2];     // B,1,S
  const int*   ext_idx  = (const int*)d_in[3];     // B,S
  const float* fc_w     = (const float*)d_in[5];   // V,D
  const float* fc_b     = (const float*)d_in[6];   // V
  const float* pgen_w   = (const float*)d_in[7];   // 1,D
  const float* pgen_b   = (const float*)d_in[8];   // 1
  const float* wq       = (const float*)d_in[9];   // D,D
  const float* bq       = (const float*)d_in[10];  // D
  const float* wk       = (const float*)d_in[11];  // D,D
  const float* bk       = (const float*)d_in[12];  // D
  float* out = (float*)d_out;

  char* ws = (char*)d_ws;
  size_t off = 0;
  auto take = [&](size_t bytes) -> char* {
    char* p = ws + off;
    off = (off + bytes + 255) & ~(size_t)255;
    return p;
  };
  unsigned short* fcw_bf = (unsigned short*)take((size_t)CV * CD * 2);   // 51.2 MB
  unsigned short* x_bf   = (unsigned short*)take((size_t)CM * CD * 2);
  unsigned short* enc_bf = (unsigned short*)take((size_t)CKR * CD * 2);
  unsigned short* wq_bf  = (unsigned short*)take((size_t)CD * CD * 2);
  unsigned short* wk_bf  = (unsigned short*)take((size_t)CD * CD * 2);
  unsigned short* q_bf   = (unsigned short*)take((size_t)CM * CD * 2);
  unsigned short* k_bf   = (unsigned short*)take((size_t)CKR * CD * 2);
  float* pgen   = (float*)take((size_t)CM * 4);
  float* sumexp = (float*)take((size_t)CM * 4);
  float* attn   = (float*)take((size_t)CM * CS * 4);                     // 1.6 MB

  // 1) bf16 conversions
  f32_to_bf16_kernel<<<8192, 256, 0, stream>>>(fc_w, fcw_bf, CV * CD);
  f32_to_bf16_kernel<<<512, 256, 0, stream>>>(x, x_bf, CM * CD);
  f32_to_bf16_kernel<<<512, 256, 0, stream>>>(enc, enc_bf, CKR * CD);
  f32_to_bf16_kernel<<<256, 256, 0, stream>>>(wq, wq_bf, CD * CD);
  f32_to_bf16_kernel<<<256, 256, 0, stream>>>(wk, wk_bf, CD * CD);
  fill_kernel<<<4, 256, 0, stream>>>(sumexp, CM, 0.0f);

  // 2) p_gen
  pgen_kernel<<<128, 256, 0, stream>>>(x, pgen_w, pgen_b, pgen);

  // 3) q/k projections (WMMA)
  {
    dim3 g(CD / 128, (CM + CKR) / 64);   // 4 x 41
    qk_gemm<<<g, 256, 0, stream>>>(x_bf, enc_bf, wq_bf, wk_bf, bq, bk, q_bf, k_bf);
  }

  // 4) attention scores + softmax + attn_dist (WMMA)
  attn_kernel<<<CB * (CT / 16), 256, 0, stream>>>(q_bf, k_bf, src_mask, pgen, attn);

  // 5) vocab GEMM pass 1: per-row sum of exp(logit)
  {
    dim3 g((CV + 127) / 128, CM / 128);  // 391 x 8
    vocab_gemm<1><<<g, 256, 0, stream>>>(x_bf, fcw_bf, fc_b, pgen, sumexp, out);
  }

  // 6) vocab GEMM pass 2: out = p_gen * softmax(logits)
  {
    dim3 g((CV + 127) / 128, CM / 128);
    vocab_gemm<2><<<g, 256, 0, stream>>>(x_bf, fcw_bf, fc_b, pgen, sumexp, out);
  }

  // 7) zero extended-vocab tail, then scatter-add attention distribution
  tailzero_kernel<<<(CM * CNX + 255) / 256, 256, 0, stream>>>(out);
  scatter_kernel<<<(CM * CS + 255) / 256, 256, 0, stream>>>(ext_idx, attn, out);

  // 8) final log
  log_kernel<<<8192, 256, 0, stream>>>(out, (long long)CM * CVEXT);
}
